// AttnShaper_4157528343030
// MI455X (gfx1250) — matrix operand order repaired and checked by reading, not compiled, not run on hardware
//
#include <hip/hip_runtime.h>

// ---------------------------------------------------------------------------
// AttnShaper for MI455X (gfx1250, wave32, WMMA)
//
// Phase 1 (parallel, WMMA): scores[t][b][p] =
//   clip( relu( conv1d(x, w, stride=8, K=32) + bias ) @ keys, 0, 6 )
// Both contractions are K=32 -> one v_wmma_f32_16x16x32_f16 each.
// 4 waves per block, one 16(t) x 16(p) tile per wave.
//
// Phase 2 (serial scan, wave32): per-batch chain over T=16384 steps:
//   prob = softmax(score - avg); avg += prob - 1/16;
//   out[b, t*8+w] = relu(dot(prob, shapes[:,w]) - x[b, t*8+w])
// One wave carries two batch chains (16 lanes each); reductions are 4-stage
// shfl_xor butterflies inside the 16-lane half-wave. Score/x loads are
// software-pipelined one step ahead (plus global_prefetch_b8 further out)
// so only ALU/shuffle latency remains on the serial chain.
// ---------------------------------------------------------------------------

#define HIST   32
#define WIN    8
#define NPAT   16
#define DIM    32
#define BATCH  128
#define LEN    131072
#define TSTEPS (LEN / WIN)   // 16384
#define TILES_PER_BLK 4

typedef _Float16 v16h __attribute__((ext_vector_type(16)));
typedef float    v8f  __attribute__((ext_vector_type(8)));

// ---------------------------------------------------------------------------
// Kernel 1: conv + relu + keys-GEMM + clip -> f16 scores[t][b][p]
// ---------------------------------------------------------------------------
__global__ __launch_bounds__(32 * TILES_PER_BLK)
void attnshaper_scores_kernel(const float* __restrict__ x,
                              const float* __restrict__ conv_w,
                              const float* __restrict__ conv_b,
                              const float* __restrict__ keys,
                              _Float16* __restrict__ scores)
{
    const int wv   = threadIdx.x >> 5;                 // wave in block
    const int lane = threadIdx.x & 31;
    const int tile = blockIdx.x * TILES_PER_BLK + wv;  // 0 .. T/16-1
    const int b    = blockIdx.y;                       // 0 .. 127
    const int t0   = tile * 16;

    __shared__ float    xs[TILES_PER_BLK][160];        // x window (152 used)
    __shared__ _Float16 outt[TILES_PER_BLK][16 * DIM]; // relu(conv+bias) tile

    // Stage x[b, t0*8-31 .. t0*8+120] into LDS (left zero padding).
    for (int i = lane; i < 152; i += 32) {
        int col = t0 * WIN - (HIST - 1) + i;
        float v = 0.0f;
        if (col >= 0 && col < LEN) v = x[(long)b * LEN + col];
        xs[wv][i] = v;
    }
    __syncthreads();

    const int m    = lane & 15;          // row (local t) / col (n) index
    const int hi   = lane >> 4;          // half-wave selector
    const int koff = hi ? 16 : 0;        // unused; B-operand K offset for this half is hi*8

    // A-operand (16x32 f16): row m = local t, K = conv tap.
    // ISA layout: VGPR j<4: K = (hi?8:0)+2j ; j>=4: K = (hi?24:16)+2(j-4).
    v16h a;
#pragma unroll
    for (int j = 0; j < 8; ++j) {
        int k = (j < 4) ? ((hi ? 8 : 0) + 2 * j)
                        : ((hi ? 24 : 16) + 2 * (j - 4));
        a[2 * j]     = (_Float16)xs[wv][m * WIN + k];
        a[2 * j + 1] = (_Float16)xs[wv][m * WIN + k + 1];
    }

    // Conv as GEMM: C[16t x 16d] = A[16t x 32k] * W^T[32k x 16d], two d-tiles.
#pragma unroll
    for (int g = 0; g < 2; ++g) {
        const int d = g * 16 + m;        // output channel (B-operand column n)
        v16h wb;
#pragma unroll
        for (int j = 0; j < 8; ++j) {
            wb[2 * j]     = (_Float16)conv_w[d * HIST + hi * 8 + (j >= 4 ? 8 : 0) + 2 * j];
            wb[2 * j + 1] = (_Float16)conv_w[d * HIST + hi * 8 + (j >= 4 ? 8 : 0) + 2 * j + 1];
        }
        v8f c = {};
        c = __builtin_amdgcn_wmma_f32_16x16x32_f16(
                false, a, false, wb, (short)0, c, false, false);

        // bias + relu; C layout: VGPR r -> (t_local = r + hi*8, col = m)
        const float bias = conv_b[g * 16 + m];
#pragma unroll
        for (int r = 0; r < 8; ++r) {
            float v = c[r] + bias;
            v = v > 0.0f ? v : 0.0f;
            outt[wv][(r + hi * 8) * DIM + (g * 16 + m)] = (_Float16)v;
        }
    }
    __syncthreads();

    // Second GEMM: scores[16t x 16p] = out[16t x 32d] * keys[32d x 16p]
    v16h a2;
#pragma unroll
    for (int j = 0; j < 8; ++j) {
        int k = (j < 4) ? ((hi ? 8 : 0) + 2 * j)
                        : ((hi ? 24 : 16) + 2 * (j - 4));
        a2[2 * j]     = outt[wv][m * DIM + k];
        a2[2 * j + 1] = outt[wv][m * DIM + k + 1];
    }
    v16h kb;                              // keys[d][p], n = p = m
#pragma unroll
    for (int j = 0; j < 8; ++j) {
        kb[2 * j]     = (_Float16)keys[(hi * 8 + (j >= 4 ? 8 : 0) + 2 * j) * NPAT + m];
        kb[2 * j + 1] = (_Float16)keys[(hi * 8 + (j >= 4 ? 8 : 0) + 2 * j + 1) * NPAT + m];
    }
    v8f s = {};
    s = __builtin_amdgcn_wmma_f32_16x16x32_f16(
            false, a2, false, kb, (short)0, s, false, false);

    // relu6 clip, store f16 scores[t][b][p]
#pragma unroll
    for (int r = 0; r < 8; ++r) {
        float v = s[r];
        v = v < 0.0f ? 0.0f : (v > 6.0f ? 6.0f : v);
        int t = t0 + r + hi * 8;
        scores[((long)t * BATCH + b) * NPAT + m] = (_Float16)v;
    }
}

// ---------------------------------------------------------------------------
// Kernel 2: sequential softmax-EMA scan + shapes projection + output.
// Lanes 0-15 -> batch 2i (p = lane), lanes 16-31 -> batch 2i+1 (p = lane-16).
// ---------------------------------------------------------------------------
__global__ __launch_bounds__(32)
void attnshaper_scan_kernel(const _Float16* __restrict__ scores,
                            const float* __restrict__ x,
                            const float* __restrict__ avg_scores,
                            const float* __restrict__ shapes,
                            float* __restrict__ out)
{
    const int lane = threadIdx.x;
    const int p    = lane & 15;
    const int b    = blockIdx.x * 2 + (lane >> 4);

    float shp[WIN];
#pragma unroll
    for (int w = 0; w < WIN; ++w) shp[w] = shapes[p * WIN + w];

    float avg = avg_scores[b * NPAT + p];

    const _Float16* sp = scores + (long)b * NPAT + p;  // step stride BATCH*NPAT
    const float*    xb = x + (long)b * LEN;
    float*          ob = out + (long)b * LEN;

    // Software pipeline: score/x for step t loaded during step t-1.
    float sc_next = (float)sp[0];
    float xi_next = (p < WIN) ? xb[p] : 0.0f;

    for (int t = 0; t < TSTEPS; ++t) {
        const float sc = sc_next;
        const float xi = xi_next;

        // Issue next step's loads before touching the serial chain.
        {
            int tn = (t + 1 < TSTEPS) ? (t + 1) : t;
            sc_next = (float)sp[(long)tn * (BATCH * NPAT)];
            if (p < WIN) xi_next = xb[tn * WIN + p];
            // Stream-ahead prefetch (lowers to global_prefetch_b8).
            int tp = t + 16;
            if (tp < TSTEPS) {
                __builtin_prefetch(sp + (long)tp * (BATCH * NPAT), 0, 1);
                __builtin_prefetch(xb + tp * WIN + p, 0, 1);
            }
        }

        // --- critical chain: max -> exp -> sum -> prob -> avg update ---
        float v  = sc - avg;
        float mx = v;
#pragma unroll
        for (int mask = 1; mask <= 8; mask <<= 1)
            mx = fmaxf(mx, __shfl_xor(mx, mask, 32));

        float e = __expf(v - mx);

        float sum = e;
#pragma unroll
        for (int mask = 1; mask <= 8; mask <<= 1)
            sum += __shfl_xor(sum, mask, 32);

        float prob = __fdividef(e, sum);
        avg += prob - (1.0f / (float)NPAT);

        // --- off-chain: signal = probs . shapes, out = relu(signal - x) ---
        float sg[WIN];
#pragma unroll
        for (int w = 0; w < WIN; ++w) sg[w] = prob * shp[w];
#pragma unroll
        for (int w = 0; w < WIN; ++w) {
#pragma unroll
            for (int mask = 1; mask <= 8; mask <<= 1)
                sg[w] += __shfl_xor(sg[w], mask, 32);
        }
        if (p < WIN) {
            float sig = sg[0];
#pragma unroll
            for (int w = 1; w < WIN; ++w)
                if (p == w) sig = sg[w];
            float o = sig - xi;
            ob[t * WIN + p] = o > 0.0f ? o : 0.0f;
        }
    }
}

// ---------------------------------------------------------------------------
// Launch
// ---------------------------------------------------------------------------
extern "C" void kernel_launch(void* const* d_in, const int* in_sizes, int n_in,
                              void* d_out, int out_size, void* d_ws, size_t ws_size,
                              hipStream_t stream)
{
    (void)in_sizes; (void)n_in; (void)out_size; (void)ws_size;

    const float* x          = (const float*)d_in[0];   // [128, 131072]
    const float* avg_scores = (const float*)d_in[1];   // [128, 16]
    const float* conv_w     = (const float*)d_in[2];   // [32, 1, 32]
    const float* conv_b     = (const float*)d_in[3];   // [32]
    const float* keys       = (const float*)d_in[4];   // [32, 16]
    const float* shapes     = (const float*)d_in[5];   // [16, 8]
    float*       out        = (float*)d_out;           // [128, 131072]

    // workspace: f16 scores[T][B][P] = 16384*128*16*2 = 64 MiB
    _Float16* scores = (_Float16*)d_ws;

    dim3 g1(TSTEPS / 16 / TILES_PER_BLK, BATCH);
    attnshaper_scores_kernel<<<g1, 32 * TILES_PER_BLK, 0, stream>>>(
        x, conv_w, conv_b, keys, scores);
    attnshaper_scan_kernel<<<BATCH / 2, 32, 0, stream>>>(
        scores, x, avg_scores, shapes, out);
}
